// ArcFaceMHS_30408368455862
// MI455X (gfx1250) — compile-verified, matrix operand order checked by reading, not run on hardware
//
#include <hip/hip_runtime.h>
#include <hip/hip_bf16.h>
#include <math.h>

// ---------------------------------------------------------------------------
// ArcFace (grouped weights) loss for MI455X / gfx1250 (wave32, WMMA).
//
//   logits : (B=512, E=256) f32
//   labels : (B,) int  (harness contract: integer -> const int*)
//   weight : (G=64, C=2048, E=256) f32
//   out    : scalar f32 loss
//
// Math:  cos[b,c] = dot(W[g_b,c,:], x_b) / (max(||W[g_b,c]||,eps) * max(||x_b||,eps))
//        margin at target column, *64, log-softmax, mean NLL.
// ---------------------------------------------------------------------------

#define B_   512
#define E_   256
#define G_   64
#define C_   2048
#define NCLS (G_ * C_)

#define SCALE_   64.0f
#define COS_M_   0.8775825618903728f   // cos(0.5)
#define SIN_M_   0.479425538604203f    // sin(0.5)
#define THETA_   (-0.8775825618903728f) // cos(pi-0.5)
#define SINMM_   0.2397127693021015f   // sin(pi-0.5)*0.5
#define EPS_     1e-12f

typedef __attribute__((ext_vector_type(16))) _Float16 v16h;
typedef __attribute__((ext_vector_type(8)))  _Float16 v8h;
typedef __attribute__((ext_vector_type(8)))  float    v8f;

// ---------------------------------------------------------------------------
// Kernel 1: L2-normalize each embedding row, store as f16.
// One block (256 threads) per sample; E_ == blockDim.
// ---------------------------------------------------------------------------
__global__ __launch_bounds__(256) void k_norm_x(const float* __restrict__ logits,
                                                _Float16* __restrict__ xh) {
  __shared__ float red[256];
  const int b = blockIdx.x;
  const int t = threadIdx.x;
  const float v = logits[(size_t)b * E_ + t];
  red[t] = v * v;
  __syncthreads();
#pragma unroll
  for (int s = 128; s > 0; s >>= 1) {
    if (t < s) red[t] += red[t + s];
    __syncthreads();
  }
  const float inv = 1.0f / fmaxf(sqrtf(red[0]), EPS_);
  xh[(size_t)b * E_ + t] = (_Float16)(v * inv);
}

// ---------------------------------------------------------------------------
// Kernel 2: per-class inverse row norms of W.  One wave32 per row of E_=256:
// each lane loads 8 floats, wave reduction via __shfl_xor (wave32).
// ---------------------------------------------------------------------------
__global__ __launch_bounds__(256) void k_w_rnorm(const float* __restrict__ W,
                                                 float* __restrict__ rnorm) {
  const int wavesPerBlk = 256 / 32;
  const int row  = blockIdx.x * wavesPerBlk + (threadIdx.x >> 5);
  const int lane = threadIdx.x & 31;
  if (row >= NCLS) return;
  const float* p = W + (size_t)row * E_ + lane * 8;
  const float4 a0 = *reinterpret_cast<const float4*>(p);
  const float4 a1 = *reinterpret_cast<const float4*>(p + 4);
  float s = a0.x * a0.x + a0.y * a0.y + a0.z * a0.z + a0.w * a0.w +
            a1.x * a1.x + a1.y * a1.y + a1.z * a1.z + a1.w * a1.w;
#pragma unroll
  for (int off = 16; off > 0; off >>= 1) s += __shfl_xor(s, off, 32);
  if (lane == 0) rnorm[row] = 1.0f / fmaxf(sqrtf(s), EPS_);
}

// ---------------------------------------------------------------------------
// Kernel 3: cosine logits via V_WMMA_F32_16X16X32_F16.
// One block per sample (8 waves); each wave owns 16-class tiles.
// A = 16x32 f16 tile of W[g] (converted from f32 in regs), B = embedding
// broadcast into all 16 columns.  CDNA5 16-bit A layout:
//   lane<16  : row M=lane,    K in {k..k+7,  k+16..k+23}
//   lane>=16 : row M=lane-16, K in {k+8..k+15, k+24..k+31}
// C layout: lanes 0/16 column N=0 hold M=0..7 / M=8..15 in VGPRs 0..7.
// ---------------------------------------------------------------------------
__global__ __launch_bounds__(256) void k_cos_gemm(const float* __restrict__ W,
                                                  const _Float16* __restrict__ xh,
                                                  const float* __restrict__ rnorm,
                                                  const int* __restrict__ labels,
                                                  float* __restrict__ cosmat) {
  const int b   = blockIdx.x;
  const int lab = labels[b];
  const int g   = (lab >= 0) ? (lab / C_) : 0;

  const float*    Wg   = W      + (size_t)g * C_ * E_;
  const _Float16* xrow = xh     + (size_t)b * E_;
  const float*    rg   = rnorm  + (size_t)g * C_;
  float*          orow = cosmat + (size_t)b * C_;

  const int lane = threadIdx.x & 31;
  const int wave = threadIdx.x >> 5;
  const int half = lane >> 4;   // 0: K-chunks {0,16}, 1: K-chunks {8,24}
  const int m    = lane & 15;   // matrix row within tile

  for (int tile = wave; tile < C_ / 16; tile += 8) {
    const float* arow = Wg + (size_t)(tile * 16 + m) * E_;
    v8f acc = {};
#pragma unroll
    for (int k = 0; k < E_; k += 32) {
      const int c0 = k + half * 8;
      const int c1 = k + 16 + half * 8;
      // Stream-prefetch the next K slab of this row (global_prefetch_b8).
      __builtin_prefetch(arow + k + 32, 0, 0);

      const float4 w00 = *reinterpret_cast<const float4*>(arow + c0);
      const float4 w01 = *reinterpret_cast<const float4*>(arow + c0 + 4);
      const float4 w10 = *reinterpret_cast<const float4*>(arow + c1);
      const float4 w11 = *reinterpret_cast<const float4*>(arow + c1 + 4);

      v16h a;
      a[0]  = (_Float16)w00.x; a[1]  = (_Float16)w00.y;
      a[2]  = (_Float16)w00.z; a[3]  = (_Float16)w00.w;
      a[4]  = (_Float16)w01.x; a[5]  = (_Float16)w01.y;
      a[6]  = (_Float16)w01.z; a[7]  = (_Float16)w01.w;
      a[8]  = (_Float16)w10.x; a[9]  = (_Float16)w10.y;
      a[10] = (_Float16)w10.z; a[11] = (_Float16)w10.w;
      a[12] = (_Float16)w11.x; a[13] = (_Float16)w11.y;
      a[14] = (_Float16)w11.z; a[15] = (_Float16)w11.w;

      // B operand: same embedding in every column (K layout mirrors A).
      const v8h x0 = *reinterpret_cast<const v8h*>(xrow + c0);
      const v8h x1 = *reinterpret_cast<const v8h*>(xrow + c1);
      v16h bb;
#pragma unroll
      for (int i = 0; i < 8; ++i) { bb[i] = x0[i]; bb[8 + i] = x1[i]; }

      acc = __builtin_amdgcn_wmma_f32_16x16x32_f16(
          /*neg_a=*/false, a, /*neg_b=*/false, bb,
          /*c_mod=*/(short)0, acc, /*reuse_a=*/false, /*reuse_b=*/false);
    }
    // Column N=0 lives in lanes 0 (M=0..7) and 16 (M=8..15).
    if (m == 0) {
      const int mbase = tile * 16 + half * 8;
#pragma unroll
      for (int r = 0; r < 8; ++r) orow[mbase + r] = acc[r] * rg[mbase + r];
    }
  }
}

// ---------------------------------------------------------------------------
// Kernel 4: additive-angular margin + scaled log-softmax CE per sample.
// One block (256 threads) per sample; fixed-order LDS tree reductions.
// ---------------------------------------------------------------------------
__global__ __launch_bounds__(256) void k_margin_ce(const float* __restrict__ cosmat,
                                                   const int* __restrict__ labels,
                                                   float* __restrict__ losses) {
  __shared__ float red[256];
  const int b   = blockIdx.x;
  const int t   = threadIdx.x;
  const int lab = labels[b];
  const bool valid = (lab != -1);
  const int local  = valid ? (lab % C_) : 0;
  const float* row = cosmat + (size_t)b * C_;

  const float tv     = row[local];
  const float sin_t  = sqrtf(fmaxf(1.0f - tv * tv, 0.0f));
  const float cos_tm = tv * COS_M_ - sin_t * SIN_M_;
  float ft = (tv > THETA_) ? cos_tm : (tv - SINMM_);
  ft = valid ? ft : tv;

  // pass 1: max of scaled logits
  float mx = -INFINITY;
  for (int c = t; c < C_; c += 256) {
    const float v = ((c == local) ? ft : row[c]) * SCALE_;
    mx = fmaxf(mx, v);
  }
  red[t] = mx;
  __syncthreads();
#pragma unroll
  for (int s = 128; s > 0; s >>= 1) {
    if (t < s) red[t] = fmaxf(red[t], red[t + s]);
    __syncthreads();
  }
  mx = red[0];
  __syncthreads();

  // pass 2: sum of exp
  float se = 0.0f;
  for (int c = t; c < C_; c += 256) {
    const float v = ((c == local) ? ft : row[c]) * SCALE_;
    se += expf(v - mx);
  }
  red[t] = se;
  __syncthreads();
#pragma unroll
  for (int s = 128; s > 0; s >>= 1) {
    if (t < s) red[t] += red[t + s];
    __syncthreads();
  }
  if (t == 0) losses[b] = -((ft * SCALE_ - mx) - logf(red[0]));
}

// ---------------------------------------------------------------------------
// Kernel 5: mean over B per-sample losses (fixed-order tree).
// ---------------------------------------------------------------------------
__global__ __launch_bounds__(512) void k_mean(const float* __restrict__ losses,
                                              float* __restrict__ out) {
  __shared__ float red[512];
  const int t = threadIdx.x;
  red[t] = losses[t];
  __syncthreads();
#pragma unroll
  for (int s = 256; s > 0; s >>= 1) {
    if (t < s) red[t] += red[t + s];
    __syncthreads();
  }
  if (t == 0) out[0] = red[0] * (1.0f / (float)B_);
}

// ---------------------------------------------------------------------------
extern "C" void kernel_launch(void* const* d_in, const int* in_sizes, int n_in,
                              void* d_out, int out_size, void* d_ws, size_t ws_size,
                              hipStream_t stream) {
  const float* logits = (const float*)d_in[0];
  const int*   labels = (const int*)d_in[1];
  const float* weight = (const float*)d_in[2];
  float*       out    = (float*)d_out;

  // workspace layout (all 256B-aligned sections)
  char* ws = (char*)d_ws;
  const size_t XH_BYTES = (size_t)B_ * E_ * sizeof(_Float16);   // 256 KB
  const size_t RN_BYTES = (size_t)NCLS * sizeof(float);         // 512 KB
  const size_t CS_BYTES = (size_t)B_ * C_ * sizeof(float);      // 4 MB
  _Float16* xh     = (_Float16*)(ws);
  float*    rnorm  = (float*)(ws + XH_BYTES);
  float*    cosmat = (float*)(ws + XH_BYTES + RN_BYTES);
  float*    losses = (float*)(ws + XH_BYTES + RN_BYTES + CS_BYTES);

  k_norm_x  <<<B_,         256, 0, stream>>>(logits, xh);
  k_w_rnorm <<<NCLS / 8,   256, 0, stream>>>(weight, rnorm);
  k_cos_gemm<<<B_,         256, 0, stream>>>(weight, xh, rnorm, labels, cosmat);
  k_margin_ce<<<B_,        256, 0, stream>>>(cosmat, labels, losses);
  k_mean    <<<1,          512, 0, stream>>>(losses, out);
}
